// AttnSlot_7902739825158
// MI455X (gfx1250) — compile-verified, matrix operand order checked by reading, not run on hardware
//
#include <hip/hip_runtime.h>
#include <hip/hip_bf16.h>
#include <stdint.h>

#define BB 16
#define QQ 512
#define LL 4096
#define DD 256
#define LT 32
#define WAVES 8
#define NEGV -1.0e12f

typedef __bf16 v16bf __attribute__((ext_vector_type(16)));
typedef float  v8f   __attribute__((ext_vector_type(8)));

union Frag { uint4 u[2]; v16bf v; };

// CDNA5 async DMA: per-lane 16B copy global -> LDS, tracked by ASYNCcnt.
__device__ __forceinline__ void async_load_b128(uint32_t lds_off, const void* gptr) {
    asm volatile("global_load_async_to_lds_b128 %0, %1, off"
                 :: "v"(lds_off), "v"((uint64_t)(uintptr_t)gptr)
                 : "memory");
}
__device__ __forceinline__ void wait_async0() {
    asm volatile("s_wait_asynccnt 0" ::: "memory");
}

__global__ __launch_bounds__(256)
void attn_flash_kernel(const float* __restrict__ hidden,
                       const float* __restrict__ outputs,
                       const unsigned char* __restrict__ mask,
                       float* __restrict__ out)
{
    // double-buffered raw f32 staging tile, filled by async DMA         (64 KB)
    __shared__ __align__(16) float  stage[2][LT * DD];
    // outputs tile bf16 row-major [l][d]  (scores-GEMM B, K=d contig)   (16 KB)
    __shared__ __align__(16) __bf16 ko_row[LT][DD];
    // outputs tile bf16 transposed [d][l] padded to 40 (PV-GEMM B)      (20 KB)
    __shared__ __align__(16) __bf16 ko_col[DD][40];
    // per-wave P staging (C-layout -> A-layout conversion)              ( 8 KB)
    __shared__ __align__(16) __bf16 pbuf[WAVES][16][LT];

    const int b    = blockIdx.y;
    const int tid  = threadIdx.x;
    const int wave = tid >> 5;
    const int lane = tid & 31;
    const int half = lane >> 4;
    const int n16  = lane & 15;
    const int kbase = half * 8;          // per-lane K base, 16-bit A/B fragment layout
    const int qw   = blockIdx.x * (WAVES * 16) + wave * 16;

    v8f ctx[16];
    #pragma unroll
    for (int t = 0; t < 16; ++t) ctx[t] = {};
    float mrow[8], lsum[8];
    #pragma unroll
    for (int r = 0; r < 8; ++r) { mrow[r] = -INFINITY; lsum[r] = 0.0f; }

    const float* hrow  = hidden + ((size_t)b * QQ + qw + n16) * DD;
    const unsigned char* mbase = mask + ((size_t)b * QQ) * (size_t)LL;
    const float* obase = outputs + (size_t)b * LL * DD;

    // ---- prologue 1: async-fetch tile 0 into stage[0] (8 x b128 per thread) ----
    {
        const uint32_t lbase = (uint32_t)(uintptr_t)(&stage[0][0]);
        #pragma unroll
        for (int it = 0; it < 8; ++it) {
            const int f4 = it * 256 + tid;                 // float4 index in tile
            async_load_b128(lbase + f4 * 16, obase + f4 * 4);
        }
    }

    // ---- prologue 2: hoist hidden A-fragments (loop-invariant), f32 -> bf16 once ----
    Frag Af[8];                                            // 8 x 16x32 fragments, 64 VGPRs
    #pragma unroll
    for (int kc = 0; kc < 8; ++kc) {
        const float* ap = hrow + kc * 32 + kbase;
        float4 a0 = *reinterpret_cast<const float4*>(ap);
        float4 a1 = *reinterpret_cast<const float4*>(ap + 4);
        float4 a2 = *reinterpret_cast<const float4*>(ap + 16);
        float4 a3 = *reinterpret_cast<const float4*>(ap + 20);
        v16bf A;
        A[0]=(__bf16)a0.x;  A[1]=(__bf16)a0.y;  A[2]=(__bf16)a0.z;  A[3]=(__bf16)a0.w;
        A[4]=(__bf16)a1.x;  A[5]=(__bf16)a1.y;  A[6]=(__bf16)a1.z;  A[7]=(__bf16)a1.w;
        A[8]=(__bf16)a2.x;  A[9]=(__bf16)a2.y;  A[10]=(__bf16)a2.z; A[11]=(__bf16)a2.w;
        A[12]=(__bf16)a3.x; A[13]=(__bf16)a3.y; A[14]=(__bf16)a3.z; A[15]=(__bf16)a3.w;
        Af[kc].v = A;
    }

    for (int l0 = 0; l0 < LL; l0 += LT) {
        const int buf = (l0 / LT) & 1;
        wait_async0();          // this wave's DMA into stage[buf] done
        __syncthreads();        // all waves' DMA done; prev compute done reading ko_*

        // ---- convert pass: stage[buf] f32 -> bf16 in both layouts ----
        #pragma unroll
        for (int it = 0; it < 8; ++it) {
            const int f4 = it * 256 + tid;
            const int l  = f4 >> 6;                        // 64 float4 per row
            const int d4 = (f4 & 63) << 2;
            const float4 v = *reinterpret_cast<const float4*>(&stage[buf][f4 * 4]);
            __bf16 e0 = (__bf16)v.x, e1 = (__bf16)v.y, e2 = (__bf16)v.z, e3 = (__bf16)v.w;
            ko_row[l][d4 + 0] = e0; ko_row[l][d4 + 1] = e1;
            ko_row[l][d4 + 2] = e2; ko_row[l][d4 + 3] = e3;
            ko_col[d4 + 0][l] = e0; ko_col[d4 + 1][l] = e1;
            ko_col[d4 + 2][l] = e2; ko_col[d4 + 3][l] = e3;
        }

        // ---- kick off DMA for the next tile into the other buffer (no wait) ----
        if (l0 + LT < LL) {
            const uint32_t lbase = (uint32_t)(uintptr_t)(&stage[buf ^ 1][0]);
            const float* gsrc = obase + (size_t)(l0 + LT) * DD;
            #pragma unroll
            for (int it = 0; it < 8; ++it) {
                const int f4 = it * 256 + tid;
                async_load_b128(lbase + f4 * 16, gsrc + f4 * 4);
            }
        }
        __syncthreads();        // bf16 tiles ready for all waves

        // ---- scores: S[16q x 32l] = Hq(16xD) * Ko^T(D x LT), pure LDS + WMMA ----
        v8f S0 = {}, S1 = {};
        #pragma unroll
        for (int kc = 0; kc < 8; ++kc) {
            Frag Bf0, Bf1;
            const __bf16* bp0 = &ko_row[n16][kc * 32 + kbase];
            Bf0.u[0] = *reinterpret_cast<const uint4*>(bp0);
            Bf0.u[1] = *reinterpret_cast<const uint4*>(bp0 + 16);
            const __bf16* bp1 = &ko_row[16 + n16][kc * 32 + kbase];
            Bf1.u[0] = *reinterpret_cast<const uint4*>(bp1);
            Bf1.u[1] = *reinterpret_cast<const uint4*>(bp1 + 16);

            S0 = __builtin_amdgcn_wmma_f32_16x16x32_bf16(false, Af[kc].v, false, Bf0.v, (short)0, S0, false, false);
            S1 = __builtin_amdgcn_wmma_f32_16x16x32_bf16(false, Af[kc].v, false, Bf1.v, (short)0, S1, false, false);
        }

        // ---- mask + online softmax ----
        float scale[8];
        #pragma unroll
        for (int r = 0; r < 8; ++r) {
            const int q = qw + half * 8 + r;               // C-layout: M = r + 8*half
            const unsigned char* mp = mbase + (size_t)q * LL + l0;
            if (mp[n16])      S0[r] = NEGV;
            if (mp[16 + n16]) S1[r] = NEGV;
            float v = fmaxf(S0[r], S1[r]);
            v = fmaxf(v, __shfl_xor(v, 1, 32));
            v = fmaxf(v, __shfl_xor(v, 2, 32));
            v = fmaxf(v, __shfl_xor(v, 4, 32));
            v = fmaxf(v, __shfl_xor(v, 8, 32));
            const float mnew = fmaxf(mrow[r], v);
            scale[r] = __expf(mrow[r] - mnew);
            mrow[r] = mnew;
            S0[r] = __expf(S0[r] - mnew);
            S1[r] = __expf(S1[r] - mnew);
            float rs = S0[r] + S1[r];
            rs += __shfl_xor(rs, 1, 32);
            rs += __shfl_xor(rs, 2, 32);
            rs += __shfl_xor(rs, 4, 32);
            rs += __shfl_xor(rs, 8, 32);
            lsum[r] = lsum[r] * scale[r] + rs;
        }
        #pragma unroll
        for (int t = 0; t < 16; ++t) {
            #pragma unroll
            for (int r = 0; r < 8; ++r) ctx[t][r] *= scale[r];
        }

        // ---- stage P bf16 (C-layout -> A-layout) through per-wave LDS ----
        #pragma unroll
        for (int r = 0; r < 8; ++r) {
            pbuf[wave][half * 8 + r][n16]      = (__bf16)S0[r];
            pbuf[wave][half * 8 + r][16 + n16] = (__bf16)S1[r];
        }
        asm volatile("s_wait_dscnt 0" ::: "memory");       // intra-wave DS ordering
        Frag Pf;
        const __bf16* pp = &pbuf[wave][n16][kbase];
        Pf.u[0] = *reinterpret_cast<const uint4*>(pp);
        Pf.u[1] = *reinterpret_cast<const uint4*>(pp + 16);

        // ---- ctx += P(16xLT) * Ko(LT x D) ----
        #pragma unroll
        for (int t = 0; t < 16; ++t) {
            Frag Vf;
            const __bf16* vp = &ko_col[t * 16 + n16][kbase];
            Vf.u[0] = *reinterpret_cast<const uint4*>(vp);
            Vf.u[1] = *reinterpret_cast<const uint4*>(vp + 16);
            ctx[t] = __builtin_amdgcn_wmma_f32_16x16x32_bf16(false, Pf.v, false, Vf.v, (short)0, ctx[t], false, false);
        }
    }

    // ---- normalize + coalesced store ----
    #pragma unroll
    for (int t = 0; t < 16; ++t) {
        #pragma unroll
        for (int r = 0; r < 8; ++r) {
            const int q = qw + half * 8 + r;
            out[((size_t)b * QQ + q) * DD + t * 16 + n16] = ctx[t][r] / lsum[r];
        }
    }
}

extern "C" void kernel_launch(void* const* d_in, const int* in_sizes, int n_in,
                              void* d_out, int out_size, void* d_ws, size_t ws_size,
                              hipStream_t stream) {
    (void)in_sizes; (void)n_in; (void)out_size; (void)d_ws; (void)ws_size;
    const float* hidden  = (const float*)d_in[0];
    const float* outputs = (const float*)d_in[1];
    const unsigned char* mask = (const unsigned char*)d_in[2];  // jax bool -> 1 byte
    float* out = (float*)d_out;

    dim3 grid(QQ / (WAVES * 16), BB);   // (4, 16)
    attn_flash_kernel<<<grid, 256, 0, stream>>>(hidden, outputs, mask, out);
}
